// ImprovedEEGGraphAttentionNetwork_30004641530338
// MI455X (gfx1250) — matrix operand order repaired and checked by reading, not compiled, not run on hardware
//
#include <hip/hip_runtime.h>
#include <hip/hip_bf16.h>
#include <math.h>

// ---------------------------------------------------------------------------
// Types for CDNA5 WMMA (wave32). 16x16x32 f16 -> f32 accumulate.
// ---------------------------------------------------------------------------
typedef __attribute__((ext_vector_type(16))) _Float16 v16h;
typedef __attribute__((ext_vector_type(8)))  _Float16 v8h;
typedef __attribute__((ext_vector_type(8)))  float    v8f;

// ---------------------------------------------------------------------------
// GEMM: C[M,Nc] = A[M,K] @ W[K,Nc]   (fp32 in, f16 WMMA, fp32 out)
// Block = 128 threads = 4 waves; each wave computes a 16x32 output tile
// (two 16x16 WMMA tiles sharing one A fragment). Block tile = 64 x 32.
// W tile staged in LDS as f16, column-major with padded stride (K+16).
// ---------------------------------------------------------------------------
template <bool SAN>
__global__ void gemm_wmma_f16(const float* __restrict__ A,
                              const float* __restrict__ W,
                              float* __restrict__ C,
                              int M, int K, int Nc)
{
    extern __shared__ _Float16 lw[];          // 32 cols x (K+16) padded
    const int KP    = K + 16;
    const int ntile = blockIdx.x * 32;
    const int mbase = blockIdx.y * 64;
    const int tid   = threadIdx.x;

    // stage W tile transposed: lw[j*KP + k] = W[k, ntile+j], coalesced in j
    for (int idx = tid; idx < K * 32; idx += blockDim.x) {
        int k = idx >> 5, j = idx & 31;
        lw[(size_t)j * KP + k] = (_Float16)W[(size_t)k * Nc + ntile + j];
    }
    __syncthreads();

    const int wave  = tid >> 5;
    const int lane  = tid & 31;
    const int mtile = mbase + wave * 16;
    const int row   = mtile + (lane & 15);
    const bool rvalid = (row < M);
    const float* arow = A + (size_t)(rvalid ? row : 0) * K;  // clamped; garbage rows never stored
    const int kA = (lane >> 4) * 16;          // A: lane-group K offset
    const int nB = lane & 15;                 // B: column within 16-wide tile
    const int kB = (lane >> 4) * 16;          // B: lane-group K offset
    const _Float16* bcol0 = lw + (size_t)nB * KP + kB;
    const _Float16* bcol1 = lw + (size_t)(nB + 16) * KP + kB;

    v8f acc0 = {}, acc1 = {};
#pragma unroll 4
    for (int k0 = 0; k0 < K; k0 += 32) {
        __builtin_prefetch(arow + k0 + 128, 0, 0);   // speculative, OOB dropped

        const float4 x0 = *(const float4*)(arow + k0 + kA);
        const float4 x1 = *(const float4*)(arow + k0 + kA + 4);
        const float4 x2 = *(const float4*)(arow + k0 + kA + 8);
        const float4 x3 = *(const float4*)(arow + k0 + kA + 12);

        float af[16] = { x0.x, x0.y, x0.z, x0.w, x1.x, x1.y, x1.z, x1.w,
                         x2.x, x2.y, x2.z, x2.w, x3.x, x3.y, x3.z, x3.w };
        v16h a;
#pragma unroll
        for (int e = 0; e < 16; ++e) {
            float v = af[e];
            if (SAN && !(v == v)) v = 0.0f;   // nan_to_num for the input layer
            a[e] = (_Float16)v;
        }

        const v8h b0lo = *(const v8h*)(bcol0 + k0);
        const v8h b0hi = *(const v8h*)(bcol0 + k0 + 8);
        const v8h b1lo = *(const v8h*)(bcol1 + k0);
        const v8h b1hi = *(const v8h*)(bcol1 + k0 + 8);
        v16h b0, b1;
#pragma unroll
        for (int e = 0; e < 8; ++e) {
            b0[e] = b0lo[e]; b0[e + 8] = b0hi[e];
            b1[e] = b1lo[e]; b1[e + 8] = b1hi[e];
        }

        acc0 = __builtin_amdgcn_wmma_f32_16x16x32_f16(
                   false, a, false, b0, (short)0, acc0, false, false);
        acc1 = __builtin_amdgcn_wmma_f32_16x16x32_f16(
                   false, a, false, b1, (short)0, acc1, false, false);
    }

    // C/D layout: lane L -> n = L%16, rows m = mtile + (L/16)*8 + r
    const int n0    = ntile + (lane & 15);
    const int mrow0 = mtile + (lane >> 4) * 8;
#pragma unroll
    for (int r = 0; r < 8; ++r) {
        int m = mrow0 + r;
        if (m < M) {
            C[(size_t)m * Nc + n0]      = acc0[r];
            C[(size_t)m * Nc + n0 + 16] = acc1[r];
        }
    }
}

// ---------------------------------------------------------------------------
// Helpers / elementwise kernels
// ---------------------------------------------------------------------------
__device__ inline void atomicMaxF(float* a, float v) {
    // sign-folded integer atomic max on IEEE floats
    if (v >= 0.0f) atomicMax((int*)a, __float_as_int(v));
    else           atomicMin((unsigned int*)a, (unsigned int)__float_as_int(v));
}

__global__ void fill_f32(float* __restrict__ p, float v, int n) {
    int i = blockIdx.x * blockDim.x + threadIdx.x;
    if (i < n) p[i] = v;
}

__global__ void init_bias_rows(float* __restrict__ out, const float* __restrict__ bias,
                               int N, int D) {
    int i = blockIdx.x * blockDim.x + threadIdx.x;
    if (i < N * D) out[i] = bias[i % D];
}

// a_s[n,h] = sum_c h[n,h,c]*a_src[h,c] ; same for a_d
__global__ void attn_proj(const float* __restrict__ h,
                          const float* __restrict__ a_src,
                          const float* __restrict__ a_dst,
                          float* __restrict__ as_, float* __restrict__ ad_,
                          int N, int H, int C) {
    int i = blockIdx.x * blockDim.x + threadIdx.x;
    if (i >= N * H) return;
    int n = i / H, hh = i % H;
    const float* hp = h + (size_t)n * H * C + (size_t)hh * C;
    float s = 0.f, d = 0.f;
    for (int c = 0; c < C; ++c) {
        float v = hp[c];
        s += v * a_src[hh * C + c];
        d += v * a_dst[hh * C + c];
    }
    as_[i] = s; ad_[i] = d;
}

// e = leaky_relu(a_s[src]+a_d[dst]); segment max over dst
__global__ void edge_logits(const int* __restrict__ ei, int E, int N, int H,
                            const float* __restrict__ as_,
                            const float* __restrict__ ad_,
                            float* __restrict__ ew, float* __restrict__ mbuf) {
    int eid = blockIdx.x * blockDim.x + threadIdx.x;
    if (eid >= E + N) return;
    int s, d;
    if (eid < E) { s = ei[eid]; d = ei[E + eid]; } else { s = d = eid - E; }
    for (int h = 0; h < H; ++h) {
        float e = as_[s * H + h] + ad_[d * H + h];
        e = (e > 0.f) ? e : 0.2f * e;
        ew[(size_t)eid * H + h] = e;
        atomicMaxF(&mbuf[d * H + h], e);
    }
}

// ex = exp(e - m[dst]); segment sum over dst (ew updated in place)
__global__ void edge_exp(const int* __restrict__ ei, int E, int N, int H,
                         float* __restrict__ ew,
                         const float* __restrict__ mbuf,
                         float* __restrict__ denom) {
    int eid = blockIdx.x * blockDim.x + threadIdx.x;
    if (eid >= E + N) return;
    int d;
    if (eid < E) { d = ei[E + eid]; } else { d = eid - E; }
    for (int h = 0; h < H; ++h) {
        float m = mbuf[d * H + h];
        if (m < -1e29f) m = 0.f;                    // where(isfinite(m), m, 0)
        float v = __expf(ew[(size_t)eid * H + h] - m);
        ew[(size_t)eid * H + h] = v;
        atomicAdd(&denom[d * H + h], v);
    }
}

// out[dst] += h[src] * alpha  (one block per edge)
__global__ void edge_aggregate(const int* __restrict__ ei, int E, int N, int H, int C,
                               const float* __restrict__ hfeat,
                               const float* __restrict__ ex,
                               const float* __restrict__ denom,
                               float* __restrict__ out) {
    int eid = blockIdx.x;
    int s, d;
    if (eid < E) { s = ei[eid]; d = ei[E + eid]; } else { s = d = eid - E; }
    int HC = H * C;
    for (int f = threadIdx.x; f < HC; f += blockDim.x) {
        int h = f / C;
        float alpha = ex[(size_t)eid * H + h] / (denom[d * H + h] + 1e-16f);
        atomicAdd(&out[(size_t)d * HC + f], hfeat[(size_t)s * HC + f] * alpha);
    }
}

// per-column mean / rsqrt(var+eps)   (one block per column)
__global__ void bn_stats(const float* __restrict__ x, int N, int D,
                         float* __restrict__ mu, float* __restrict__ rsig) {
    __shared__ float ssum[256], ssq[256];
    int c = blockIdx.x, tid = threadIdx.x;
    float s = 0.f, q = 0.f;
    for (int n = tid; n < N; n += blockDim.x) {
        float v = x[(size_t)n * D + c];
        s += v; q += v * v;
    }
    ssum[tid] = s; ssq[tid] = q; __syncthreads();
    for (int off = 128; off > 0; off >>= 1) {
        if (tid < off) { ssum[tid] += ssum[tid + off]; ssq[tid] += ssq[tid + off]; }
        __syncthreads();
    }
    if (tid == 0) {
        float m = ssum[0] / (float)N;
        float var = ssq[0] / (float)N - m * m;
        mu[c] = m; rsig[c] = rsqrtf(var + 1e-5f);
    }
}

__global__ void bn_elu(const float* __restrict__ x,
                       const float* __restrict__ mu, const float* __restrict__ rsig,
                       const float* __restrict__ g, const float* __restrict__ b,
                       float* __restrict__ y, int N, int D) {
    int i = blockIdx.x * blockDim.x + threadIdx.x;
    if (i >= N * D) return;
    int c = i % D;
    float v = (x[i] - mu[c]) * rsig[c] * g[c] + b[c];
    y[i] = (v > 0.f) ? v : (__expf(v) - 1.f);
}

// ---------------------------------------------------------------------------
// Pooling & head
// ---------------------------------------------------------------------------
__global__ void pool_accum(const float* __restrict__ h, const int* __restrict__ batch,
                           int N, int D, float* __restrict__ addp,
                           float* __restrict__ maxp, float* __restrict__ cnt) {
    int i = blockIdx.x * blockDim.x + threadIdx.x;
    if (i >= N * D) return;
    int n = i / D, f = i % D;
    int g = batch[n];
    float v = h[i];
    atomicAdd(&addp[g * D + f], v);
    atomicMaxF(&maxp[g * D + f], v);
    if (f == 0) atomicAdd(&cnt[g], 1.f);
}

__global__ void pool_final(const float* __restrict__ addp, const float* __restrict__ maxp,
                           const float* __restrict__ cnt, float* __restrict__ z,
                           int G, int D) {
    int i = blockIdx.x * blockDim.x + threadIdx.x;
    if (i >= G * D) return;
    int g = i / D, f = i % D;
    float a = addp[g * D + f];
    float m = maxp[g * D + f];
    if (m < -1e29f) m = 0.f;
    float c = cnt[g];
    z[(size_t)g * 3 * D + f]         = a / fmaxf(c, 1.f);
    z[(size_t)g * 3 * D + D + f]     = m;
    z[(size_t)g * 3 * D + 2 * D + f] = a;
}

__global__ void fc_small(const float* __restrict__ x, const float* __restrict__ W,
                         const float* __restrict__ b, float* __restrict__ y,
                         int M, int K, int Nc) {
    int i = blockIdx.x * blockDim.x + threadIdx.x;
    if (i >= M * Nc) return;
    int m = i / Nc, n = i % Nc;
    float s = b[n];
    for (int k = 0; k < K; ++k) s += x[(size_t)m * K + k] * W[(size_t)k * Nc + n];
    y[i] = s;
}

// ---------------------------------------------------------------------------
// Host orchestration
// ---------------------------------------------------------------------------
extern "C" void kernel_launch(void* const* d_in, const int* in_sizes, int n_in,
                              void* d_out, int out_size, void* d_ws, size_t ws_size,
                              hipStream_t stream) {
    const float* x    = (const float*)d_in[0];
    const int*   ei   = (const int*)d_in[1];
    const int*   batch= (const int*)d_in[2];
    const int N = in_sizes[2];                 // 50000
    const int E = in_sizes[1] / 2;             // 300000
    const int G = 500;

    // params: sorted-key flatten: bn1..7{b,g}, fc1..4{W,b}, gat1..4{W,a_dst,a_src,b}
    int pi = 3;
    const float *bn_b[7], *bn_g[7];
    for (int i = 0; i < 7; ++i) { bn_b[i] = (const float*)d_in[pi++]; bn_g[i] = (const float*)d_in[pi++]; }
    const float *fc_W[4], *fc_b[4];
    for (int i = 0; i < 4; ++i) { fc_W[i] = (const float*)d_in[pi++]; fc_b[i] = (const float*)d_in[pi++]; }
    const float *gat_W[4], *gat_ad[4], *gat_as[4], *gat_b[4];
    for (int i = 0; i < 4; ++i) {
        gat_W[i]  = (const float*)d_in[pi++];
        gat_ad[i] = (const float*)d_in[pi++];
        gat_as[i] = (const float*)d_in[pi++];
        gat_b[i]  = (const float*)d_in[pi++];
    }

    // workspace carve-up
    float* wbase = (float*)d_ws;
    size_t off = 0;
    auto alloc = [&](size_t n) { float* p = wbase + off; off += (n + 63) & ~(size_t)63; return p; };
    float* bufA = alloc((size_t)N * 512);
    float* bufB = alloc((size_t)N * 512);
    float* as_  = alloc((size_t)N * 8);
    float* ad_  = alloc((size_t)N * 8);
    float* ew   = alloc((size_t)(E + N) * 8);
    float* mbuf = alloc((size_t)N * 8);
    float* den  = alloc((size_t)N * 8);
    float* mu   = alloc(512);
    float* rsig = alloc(512);
    float* addp = alloc((size_t)G * 32);
    float* maxp = alloc((size_t)G * 32);
    float* cnt  = alloc(G);
    float* z    = alloc((size_t)G * 96);
    float* t1   = alloc((size_t)G * 64);
    float* t2   = alloc((size_t)G * 32);
    float* t3   = alloc((size_t)G * 16);
    (void)ws_size; (void)n_in; (void)out_size;

    struct LD { int K, H, C; };
    const LD Ls[4] = { {128, 8, 64}, {512, 8, 64}, {512, 4, 64}, {256, 1, 32} };

    const float* act = x;
    float* P = bufA;   // GEMM output (h features)
    float* Q = bufB;   // aggregation output
    const int ET = E + N;

    for (int i = 0; i < 4; ++i) {
        const int K = Ls[i].K, H = Ls[i].H, C = Ls[i].C, D = H * C;

        dim3 gg(D / 32, (N + 63) / 64);
        size_t shm = (size_t)(K + 16) * 32 * sizeof(_Float16);
        if (i == 0)
            gemm_wmma_f16<true ><<<gg, dim3(128), shm, stream>>>(act, gat_W[i], P, N, K, D);
        else
            gemm_wmma_f16<false><<<gg, dim3(128), shm, stream>>>(act, gat_W[i], P, N, K, D);

        int NH = N * H;
        attn_proj<<<(NH + 255) / 256, 256, 0, stream>>>(P, gat_as[i], gat_ad[i], as_, ad_, N, H, C);
        fill_f32<<<(NH + 255) / 256, 256, 0, stream>>>(mbuf, -1e30f, NH);
        fill_f32<<<(NH + 255) / 256, 256, 0, stream>>>(den, 0.f, NH);
        edge_logits<<<(ET + 255) / 256, 256, 0, stream>>>(ei, E, N, H, as_, ad_, ew, mbuf);
        edge_exp<<<(ET + 255) / 256, 256, 0, stream>>>(ei, E, N, H, ew, mbuf, den);

        init_bias_rows<<<(N * D + 255) / 256, 256, 0, stream>>>(Q, gat_b[i], N, D);
        edge_aggregate<<<ET, 128, 0, stream>>>(ei, E, N, H, C, P, ew, den, Q);

        bn_stats<<<D, 256, 0, stream>>>(Q, N, D, mu, rsig);
        bn_elu<<<(N * D + 255) / 256, 256, 0, stream>>>(Q, mu, rsig, bn_g[i], bn_b[i], P, N, D);

        act = P;
        float* t = P; P = Q; Q = t;
    }

    // global pooling on act [N,32]
    fill_f32<<<(G * 32 + 255) / 256, 256, 0, stream>>>(addp, 0.f, G * 32);
    fill_f32<<<(G * 32 + 255) / 256, 256, 0, stream>>>(maxp, -1e30f, G * 32);
    fill_f32<<<(G + 255) / 256, 256, 0, stream>>>(cnt, 0.f, G);
    pool_accum<<<(N * 32 + 255) / 256, 256, 0, stream>>>(act, batch, N, 32, addp, maxp, cnt);
    pool_final<<<(G * 32 + 255) / 256, 256, 0, stream>>>(addp, maxp, cnt, z, G, 32);

    // MLP head
    fc_small<<<(G * 64 + 255) / 256, 256, 0, stream>>>(z, fc_W[0], fc_b[0], t1, G, 96, 64);
    bn_stats<<<64, 256, 0, stream>>>(t1, G, 64, mu, rsig);
    bn_elu<<<(G * 64 + 255) / 256, 256, 0, stream>>>(t1, mu, rsig, bn_g[4], bn_b[4], t1, G, 64);

    fc_small<<<(G * 32 + 255) / 256, 256, 0, stream>>>(t1, fc_W[1], fc_b[1], t2, G, 64, 32);
    bn_stats<<<32, 256, 0, stream>>>(t2, G, 32, mu, rsig);
    bn_elu<<<(G * 32 + 255) / 256, 256, 0, stream>>>(t2, mu, rsig, bn_g[5], bn_b[5], t2, G, 32);

    fc_small<<<(G * 16 + 255) / 256, 256, 0, stream>>>(t2, fc_W[2], fc_b[2], t3, G, 32, 16);
    bn_stats<<<16, 256, 0, stream>>>(t3, G, 16, mu, rsig);
    bn_elu<<<(G * 16 + 255) / 256, 256, 0, stream>>>(t3, mu, rsig, bn_g[6], bn_b[6], t3, G, 16);

    fc_small<<<(G * 2 + 255) / 256, 256, 0, stream>>>(t3, fc_W[3], fc_b[3], (float*)d_out, G, 16, 2);
}
